// InnerSoftShiftTripleModule_49847390437483
// MI455X (gfx1250) — matrix operand order repaired and checked by reading, not compiled, not run on hardware
//
#include <hip/hip_runtime.h>

// ---------------------------------------------------------------------------
// Fused "soft shift" attention for MI455X (gfx1250, wave32, WMMA f16 16x16x32)
// ---------------------------------------------------------------------------

typedef _Float16 half16 __attribute__((ext_vector_type(16)));
typedef float    float8 __attribute__((ext_vector_type(8)));

#define HW      4096
#define C2      256
#define NT      32          // N-tile (columns of attention per iteration)
#define LP      272         // Lsh pitch in f16  (row = n)   544B = 32*17 -> 32B-aligned frags
#define TP      48          // LshT pitch in f16 (row = c)   96B  = 32*3
#define PP      40          // Psh pitch in f16  (row = m)   80B  = 16*5 (16B-aligned chunks)
#define NEGINF  (-1e30f)
#define EPSN    (1e-8f)

// ---- copy former+latter straight into output channels [0,512) --------------
__global__ __launch_bounds__(256) void copy_fl_kernel(const float4* __restrict__ in,
                                                      float* __restrict__ out) {
    long i = (long)blockIdx.x * 256 + threadIdx.x;      // 8*512*1024 float4
    int  b = (int)(i >> 19);                            // 512*1024 float4 per batch
    long r = i & ((1L << 19) - 1);                      // ch*1024 + p4
    float4 v = in[i];
    ((float4*)out)[(long)b * (768 * 1024) + r] = v;
}

// ---- reciprocal L2 norms of F rows and L rows: rn[0:32768]=F, [32768:]=L ---
__global__ __launch_bounds__(256) void norm_kernel(const float* __restrict__ in,
                                                   float* __restrict__ rn) {
    int id    = blockIdx.x * 256 + threadIdx.x;         // 0..65535
    int which = id >> 15;                               // 0 = former, 1 = latter
    int bm    = id & 32767;
    int b     = bm >> 12;
    int m     = bm & 4095;
    const float* p = in + (((long)b * 512 + which * 256) << 12) + m;
    float s = 0.f;
#pragma unroll 8
    for (int c = 0; c < C2; ++c) {
        float v = p[(long)c << 12];
        s += v * v;
    }
    rn[id] = 1.0f / fmaxf(sqrtf(s), EPSN);
}

// ---- fused flash-attention kernel ------------------------------------------
__global__ __launch_bounds__(256, 1) void attn_kernel(const float* __restrict__ in,
                                                      const int*   __restrict__ flag,
                                                      const float* __restrict__ rn,
                                                      float*       __restrict__ out) {
    __shared__ __attribute__((aligned(64))) _Float16 Lsh [NT * LP];    // [n][c]  17408 B
    __shared__ __attribute__((aligned(64))) _Float16 LshT[C2 * TP];    // [c][n]  24576 B
    __shared__ __attribute__((aligned(64))) _Float16 Psh [8 * 16 * PP]; //        10240 B

    const int tid   = threadIdx.x;
    const int lane  = tid & 31;
    const int w     = tid >> 5;              // wave id 0..7
    const int b     = blockIdx.y;            // batch
    const int mwv   = (blockIdx.x << 7) + (w << 4);  // this wave's 16-row base
    const int half  = lane >> 4;
    const int l16   = lane & 15;
    const int abase = half ? 8  : 0;         // 16-bit A-frag K sub-base per half
    const int boff  = half ? 16 : 0;         // 16-bit B-frag K offset per half

    const float* rnF = rn;
    const float* rnL = rn + 32768;
    const long   inL = (((long)b * 512 + 256) << 12);   // latter base

    // A fragments of F (raw values; norms folded in later). Resident all kernel.
    half16 aF[8];
    {
        const float* Fb = in + (((long)b * 512) << 12) + (mwv + l16);
#pragma unroll
        for (int kk = 0; kk < 8; ++kk) {
#pragma unroll
            for (int j = 0; j < 16; ++j) {
                int c = kk * 32 + boff + j;
                aF[kk][j] = (_Float16)Fb[(long)c << 12];
            }
        }
    }

    // per-row (vgpr i -> row mwv + i + 8*half) softmax state + output accum
    float  rf[8], mrun[8], lrun[8];
    float8 O[16];
    const float8 f8zero = {0.f, 0.f, 0.f, 0.f, 0.f, 0.f, 0.f, 0.f};
#pragma unroll
    for (int i = 0; i < 8; ++i) {
        rf[i]   = rnF[(b << 12) + mwv + i + 8 * half];
        mrun[i] = NEGINF;
        lrun[i] = 0.f;
    }
#pragma unroll
    for (int jc = 0; jc < 16; ++jc) O[jc] = f8zero;

    // fragment base pointers inside LDS (per-lane, loop-invariant)
    const _Float16* b0base = &Lsh[l16 * LP + boff];         // GEMM1 B row (jn=0)
    const _Float16* b1base = &Lsh[(16 + l16) * LP + boff];  // GEMM1 B row (jn=1)

    for (int n0 = 0; n0 < HW; n0 += NT) {
        __syncthreads();                      // previous tile fully consumed
        // stage latter tile (f32 -> f16) into both layouts
#pragma unroll
        for (int r = 0; r < 32; ++r) {
            int idx = (r << 8) + tid;         // 0..8191
            int c = idx >> 5, n = idx & 31;
            const float* gp = &in[inL + ((long)c << 12) + n0 + n];
            float v = *gp;
            if (n0 + NT < HW) __builtin_prefetch(gp + NT, 0, 1);  // global_prefetch_b8
            _Float16 h = (_Float16)v;
            Lsh [n * LP + c] = h;
            LshT[c * TP + n] = h;
        }
        __syncthreads();

        // column-side constants for this wave's lanes
        float rl[2]; int fl[2];
#pragma unroll
        for (int jn = 0; jn < 2; ++jn) {
            int n  = n0 + jn * 16 + l16;
            rl[jn] = rnL[(b << 12) + n];
            fl[jn] = flag[n];
        }

        // GEMM1: S = F_tile(16x256) x L_tile^T(256x32), kk-outer so both jn
        // WMMAs share A; B fragments explicitly double-buffered so DS loads
        // for step kk+1 overlap the WMMAs of step kk (partial dscnt waits).
        float8 S0 = f8zero, S1 = f8zero;
        half16 bA0 = *(const half16*)(b0base);
        half16 bA1 = *(const half16*)(b1base);
#pragma unroll
        for (int kk = 0; kk < 8; ++kk) {
            int kn = ((kk + 1) & 7) * 32;
            half16 bN0 = *(const half16*)(b0base + kn);
            half16 bN1 = *(const half16*)(b1base + kn);
            S0 = __builtin_amdgcn_wmma_f32_16x16x32_f16(
                     false, aF[kk], false, bA0, (short)0, S0, false, false);
            S1 = __builtin_amdgcn_wmma_f32_16x16x32_f16(
                     false, aF[kk], false, bA1, (short)0, S1, false, false);
            bA0 = bN0;
            bA1 = bN1;
        }

        // cosine scaling + column mask + online softmax (16-lane butterflies)
#pragma unroll
        for (int i = 0; i < 8; ++i) {
            float s0 = S0[i] * rf[i] * rl[0];
            float s1 = S1[i] * rf[i] * rl[1];
            if (fl[0]) s0 = NEGINF;
            if (fl[1]) s1 = NEGINF;
            float tm = fmaxf(s0, s1);
#pragma unroll
            for (int d = 1; d < 16; d <<= 1) tm = fmaxf(tm, __shfl_xor(tm, d, 32));
            float mn   = fmaxf(mrun[i], tm);
            float corr = __expf(mrun[i] - mn);
            float p0   = __expf(s0 - mn);
            float p1   = __expf(s1 - mn);
            float rs   = p0 + p1;
#pragma unroll
            for (int d = 1; d < 16; d <<= 1) rs += __shfl_xor(rs, d, 32);
            lrun[i] = lrun[i] * corr + rs;
            mrun[i] = mn;
#pragma unroll
            for (int jc = 0; jc < 16; ++jc) O[jc][i] *= corr;
            // scatter P (C/D layout) into row-major LDS tile for A-frag reuse
            int r = i + 8 * half;
            Psh[(w * 16 + r) * PP + l16]      = (_Float16)p0;
            Psh[(w * 16 + r) * PP + 16 + l16] = (_Float16)p1;
        }
        __asm__ volatile("s_wait_dscnt 0x0" ::: "memory");  // cross-lane P RAW

        // A fragment of P (16x32)
        half16 a2;
        {
            const _Float16* pp = &Psh[(w * 16 + l16) * PP];
#pragma unroll
            for (int j = 0; j < 16; ++j)
                a2[j] = pp[boff + j];
        }

        // GEMM2: O(16x256) += P(16x32) x Lv_tile(32x256); double-buffered B
        half16 bB = *(const half16*)&LshT[l16 * TP + boff];
#pragma unroll
        for (int jc = 0; jc < 16; ++jc) {
            int jn16 = (((jc + 1) & 15) * 16 + l16) * TP + boff;
            half16 bN = *(const half16*)&LshT[jn16];
            O[jc] = __builtin_amdgcn_wmma_f32_16x16x32_f16(
                        false, a2, false, bB, (short)0, O[jc], false, false);
            bB = bN;
        }
    }

    // finalize: divide by softmax sum, apply row mask, transpose via LDS,
    // write coalesced to output channels [512,768)
    float inv_l[8];
#pragma unroll
    for (int i = 0; i < 8; ++i) {
        int m    = mwv + i + 8 * half;
        float rm = (flag[m] == 1) ? 1.f : 0.f;
        inv_l[i] = rm / lrun[i];
    }
    float* Po = (float*)&Psh[w * 16 * PP];      // 16 x 20 f32 per-wave region
    long obase = (((long)b * 768 + 512) << 12) + mwv;
#pragma unroll
    for (int jc = 0; jc < 16; ++jc) {
        __asm__ volatile("s_wait_dscnt 0x0" ::: "memory");  // WAR vs prior reads
#pragma unroll
        for (int i = 0; i < 8; ++i)
            Po[l16 * 20 + i + 8 * half] = O[jc][i] * inv_l[i];
        __asm__ volatile("s_wait_dscnt 0x0" ::: "memory");  // RAW across lanes
#pragma unroll
        for (int t = 0; t < 8; ++t) {
            int cl = half + 2 * t;
            out[obase + ((long)(jc * 16 + cl) << 12) + l16] = Po[cl * 20 + l16];
        }
    }
}

// ---------------------------------------------------------------------------
extern "C" void kernel_launch(void* const* d_in, const int* in_sizes, int n_in,
                              void* d_out, int out_size, void* d_ws, size_t ws_size,
                              hipStream_t stream) {
    const float* in   = (const float*)d_in[0];  // (8,512,64,64) f32
    // d_in[1] = mask (unused by reference), d_in[2..4] = scalars (unused)
    const int*   flag = (const int*)d_in[5];    // (4096,) i32
    float*       out  = (float*)d_out;          // (8,768,64,64) f32
    float*       rn   = (float*)d_ws;           // 65536 floats: rnF | rnL

    copy_fl_kernel<<<16384, 256, 0, stream>>>((const float4*)in, out);
    norm_kernel  <<<256,   256, 0, stream>>>(in, rn);
    attn_kernel  <<<dim3(32, 8), 256, 0, stream>>>(in, flag, rn, out);
}
